// Multihead_self_atten_48129403519363
// MI455X (gfx1250) — matrix operand order repaired and checked by reading, not compiled, not run on hardware
//
#include <hip/hip_runtime.h>
#include <hip/hip_bf16.h>
#include <math.h>

typedef __attribute__((ext_vector_type(16))) __bf16 v16bf;
typedef __attribute__((ext_vector_type(8)))  __bf16 v8bf;
typedef __attribute__((ext_vector_type(8)))  float  v8f;
typedef __attribute__((ext_vector_type(4)))  unsigned int u32x4;
typedef __attribute__((ext_vector_type(8)))  int i32x8;
typedef __attribute__((ext_vector_type(4)))  int i32x4;

#define WMMA_BF16(a, b, c) \
  __builtin_amdgcn_wmma_f32_16x16x32_bf16(false, (a), false, (b), (short)0, (c), false, false)

constexpr int B_ = 4, S_ = 2048, D_ = 1024, H_ = 16, DK_ = 64;

union V16 { v16bf v; v8bf h[2]; };

// ---------------------------------------------------------------- fp32 -> bf16
__global__ void cvt_f32_to_bf16(const float* __restrict__ in, __bf16* __restrict__ out, int n) {
  int i = blockIdx.x * blockDim.x + threadIdx.x;
  if (i < n) out[i] = (__bf16)in[i];
}

// -------------------------------------------------- TDM: 64x32 bf16 W tile -> LDS
// D# per CDNA5 ISA ch.8: group0 {count, lds_addr, global_addr|type=2},
// group1 {data_size=2B, tensor_dim0/1, tile_dim0=32, tile_dim1=64, dim0_stride},
// groups 2/3 zero (2-D tensor).
__device__ __forceinline__ void tdm_load_w_tile(const __bf16* W, int N, int K,
                                                int n0, int k0, unsigned lds_off) {
  unsigned long long ga = (unsigned long long)(size_t)(W + (size_t)n0 * K + k0);
  u32x4 g0;
  g0[0] = 1u;                                   // count=1 (valid), no gather
  g0[1] = lds_off;                              // lds_addr (bytes)
  g0[2] = (unsigned)ga;                         // global_addr[31:0]
  g0[3] = (unsigned)(ga >> 32) | 0x80000000u;   // global_addr[56:32] | type=2
  i32x8 g1;
  g1[0] = 0x00010000;                                        // data_size=1 (2 bytes)
  g1[1] = (int)((unsigned)K << 16);                          // tensor_dim0[15:0]
  g1[2] = (int)(((unsigned)K >> 16) | ((unsigned)N << 16));  // td0[31:16] | td1[15:0]
  g1[3] = (int)(((unsigned)N >> 16) | (32u << 16));          // td1[31:16] | tile_dim0=32
  g1[4] = 64;                                                // tile_dim1=64, tile_dim2=0
  g1[5] = K;                                                 // tensor_dim0_stride[31:0]
  g1[6] = 0;                                                 // stride[47:32] | td1_stride lo
  g1[7] = 0;
  i32x4 gz = {0, 0, 0, 0};
#if defined(__clang_major__) && (__clang_major__ >= 23)
  i32x8 gz8 = {0, 0, 0, 0, 0, 0, 0, 0};
  __builtin_amdgcn_tensor_load_to_lds(g0, g1, gz, gz, gz8, 0);
#else
  __builtin_amdgcn_tensor_load_to_lds(g0, g1, gz, gz, 0);
#endif
}

// ----------------------------------------------------- C = A(MxK) * W(NxK)^T
// Wave tile 32x64 (8 WMMAs / k-step); W tile staged to LDS by TDM, double-buffered.
// Block = 8 waves => 256x64 block tile.
template <typename OutT>
__global__ __launch_bounds__(256)
void gemm_xwT(const __bf16* __restrict__ A, const __bf16* __restrict__ W,
              OutT* __restrict__ C, int M, int N, int K) {
  __shared__ __bf16 wt[2][64 * 32];             // 2 x 4 KB W tiles
  const int wave = threadIdx.x >> 5;
  const int lane = threadIdx.x & 31;
  const int n16  = lane & 15;
  const int hi   = lane >> 4;
  const int m0   = blockIdx.y * 256 + wave * 32;
  const int n0   = blockIdx.x * 64;

  v8f acc[2][4] = {};
  const __bf16* arow0 = A + (size_t)(m0 + n16) * K;
  const __bf16* arow1 = arow0 + (size_t)16 * K;

  if (wave == 0) {
    tdm_load_w_tile(W, N, K, n0, 0, (unsigned)(size_t)&wt[0][0]);
    __builtin_amdgcn_s_wait_tensorcnt(0);
  }
  __syncthreads();

  int cur = 0;
  for (int k0 = 0; k0 < K; k0 += 32) {
    if (wave == 0 && k0 + 32 < K)               // prefetch next tile into other buffer
      tdm_load_w_tile(W, N, K, n0, k0 + 32, (unsigned)(size_t)&wt[cur ^ 1][0]);

    V16 a0, a1;                                 // 16-bit A 16x32 per-lane layout
    a0.v = *(const v16bf*)(arow0 + k0 + hi * 16);
    a1.v = *(const v16bf*)(arow1 + k0 + hi * 16);
#pragma unroll
    for (int t = 0; t < 4; ++t) {
      v16bf b = *(const v16bf*)&wt[cur][(t * 16 + n16) * 32 + hi * 16];
      acc[0][t] = WMMA_BF16(a0.v, b, acc[0][t]);
      acc[1][t] = WMMA_BF16(a1.v, b, acc[1][t]);
    }
    if (wave == 0) __builtin_amdgcn_s_wait_tensorcnt(0);
    __syncthreads();
    cur ^= 1;
  }
#pragma unroll
  for (int ms = 0; ms < 2; ++ms) {
#pragma unroll
    for (int t = 0; t < 4; ++t) {
#pragma unroll
      for (int r = 0; r < 8; ++r) {
        int m = m0 + ms * 16 + r + hi * 8;      // C layout: lane=n, vgpr r = row
        int n = n0 + t * 16 + n16;
        C[(size_t)m * N + n] = (OutT)acc[ms][t][r];
      }
    }
  }
}

// ---------------------------------------------------------------- RoPE (Q/K)
__global__ void rope_kernel(__bf16* __restrict__ qk, const int* __restrict__ pos, int total) {
  int i = blockIdx.x * blockDim.x + threadIdx.x;
  if (i >= total) return;                       // i over (b, s, h, j) with half=32
  int j = i & 31;
  int h = (i >> 5) & 15;
  int s = (i >> 9) & (S_ - 1);
  int b = i >> 20;
  float p   = (float)pos[s];
  float inv = powf(10000.0f, -2.0f * (float)j / 64.0f);
  float ang = p * inv;
  float c = cosf(ang), sn = sinf(ang);
  size_t base = ((size_t)(b * S_ + s)) * D_ + h * DK_ + 2 * j;
  float x0 = (float)qk[base], x1 = (float)qk[base + 1];
  qk[base]     = (__bf16)(x0 * c - x1 * sn);
  qk[base + 1] = (__bf16)(x0 * sn + x1 * c);
}

// ------------------------------------------- V (B,S,D head-major) -> Vt (B,H,dk,S)
__global__ void transpose_v(const __bf16* __restrict__ V, __bf16* __restrict__ Vt, int total) {
  int i = blockIdx.x * blockDim.x + threadIdx.x;
  if (i >= total) return;
  int c = i & (D_ - 1);
  int s = (i >> 10) & (S_ - 1);
  int b = i >> 21;
  int h = c >> 6, d = c & 63;
  Vt[(((size_t)b * H_ + h) * DK_ + d) * S_ + s] = V[i];
}

// ---------------------------------------------------------------- flash attention
// grid (S/128, H, B); 8 waves/block; each wave owns 16 queries, online softmax over 32-key blocks.
__global__ __launch_bounds__(256)
void flash_attn(const __bf16* __restrict__ Q, const __bf16* __restrict__ K,
                const __bf16* __restrict__ Vt, __bf16* __restrict__ O) {
  __shared__ __bf16 plds[8 * 16 * 32];          // per-wave 16x32 P staging tile
  const int wave = threadIdx.x >> 5;
  const int lane = threadIdx.x & 31;
  const int n16  = lane & 15;
  const int hi   = lane >> 4;
  const int h = blockIdx.y, b = blockIdx.z;
  const int qbase = blockIdx.x * 128 + wave * 16;
  const float scale = 0.125f;                   // 1/sqrt(64)

  // Q tile in A layout: two d_k chunks of 32
  const __bf16* qrow = Q + ((size_t)(b * S_ + qbase + n16)) * D_ + h * DK_;
  V16 qa[2];
#pragma unroll
  for (int c = 0; c < 2; ++c) {
    qa[c].v = *(const v16bf*)(qrow + 32 * c + hi * 16);
  }

  v8f oacc[4] = {};
  float mrow[8], lrow[8];
#pragma unroll
  for (int r = 0; r < 8; ++r) { mrow[r] = -3.0e38f; lrow[r] = 0.0f; }

  __bf16* pw = plds + wave * 512;
  const int qmax = qbase + 15;

  for (int kb = 0; kb <= qmax; kb += 32) {
    // ----- scores: two 16-key subtiles, K-dim = d_k = 64 (2 WMMAs each)
    v8f sacc[2] = {};
#pragma unroll
    for (int t = 0; t < 2; ++t) {
      const __bf16* krow = K + ((size_t)(b * S_ + kb + t * 16 + n16)) * D_ + h * DK_;
#pragma unroll
      for (int c = 0; c < 2; ++c) {
        v16bf bk = *(const v16bf*)(krow + 32 * c + hi * 16);
        sacc[t] = WMMA_BF16(qa[c].v, bk, sacc[t]);
      }
    }
    // ----- online softmax (rows live across 16-lane halves)
#pragma unroll
    for (int r = 0; r < 8; ++r) {
      int q = qbase + r + hi * 8;
      float s0 = sacc[0][r] * scale;
      float s1 = sacc[1][r] * scale;
      if (kb + n16 > q)      s0 = -3.0e38f;     // causal mask
      if (kb + 16 + n16 > q) s1 = -3.0e38f;
      float tmax = fmaxf(s0, s1);
#pragma unroll
      for (int off = 1; off < 16; off <<= 1) tmax = fmaxf(tmax, __shfl_xor(tmax, off, 32));
      float mnew = fmaxf(mrow[r], tmax);
      float p0 = __expf(s0 - mnew), p1 = __expf(s1 - mnew);
      float ps = p0 + p1;
#pragma unroll
      for (int off = 1; off < 16; off <<= 1) ps += __shfl_xor(ps, off, 32);
      float corr = __expf(mrow[r] - mnew);
      lrow[r] = lrow[r] * corr + ps;
      mrow[r] = mnew;
#pragma unroll
      for (int t4 = 0; t4 < 4; ++t4) oacc[t4][r] *= corr;
      int m = r + hi * 8;                       // C-layout -> LDS row-major
      pw[m * 32 + n16]      = (__bf16)p0;
      pw[m * 32 + 16 + n16] = (__bf16)p1;
    }
    // ----- reload P in A layout (wave-private LDS, DScnt ordering by compiler)
    V16 pa;
    pa.v = *(const v16bf*)(pw + n16 * 32 + hi * 16);
    // ----- O += P * V  (B column = Vt row = contiguous keys)
#pragma unroll
    for (int t = 0; t < 4; ++t) {
      const __bf16* vrow = Vt + (((size_t)(b * H_ + h)) * DK_ + t * 16 + n16) * S_ + kb + hi * 16;
      v16bf bv = *(const v16bf*)vrow;
      oacc[t] = WMMA_BF16(pa.v, bv, oacc[t]);
    }
  }
  // ----- normalize + store attention output (b, s, h*64+d) bf16
#pragma unroll
  for (int t = 0; t < 4; ++t) {
#pragma unroll
    for (int r = 0; r < 8; ++r) {
      int q = qbase + r + hi * 8;
      float o = oacc[t][r] / lrow[r];
      O[((size_t)(b * S_ + q)) * D_ + h * DK_ + t * 16 + n16] = (__bf16)o;
    }
  }
}

// ---------------------------------------------------------------------- launch
extern "C" void kernel_launch(void* const* d_in, const int* in_sizes, int n_in,
                              void* d_out, int out_size, void* d_ws, size_t ws_size,
                              hipStream_t stream) {
  const float* x   = (const float*)d_in[0];
  const int*   pos = (const int*)d_in[1];
  const float* wq  = (const float*)d_in[2];
  const float* wk  = (const float*)d_in[3];
  const float* wv  = (const float*)d_in[4];
  const float* wo  = (const float*)d_in[5];
  float* out = (float*)d_out;

  const size_t NX = (size_t)B_ * S_ * D_;   // 8388608
  const size_t NW = (size_t)D_ * D_;        // 1048576
  char* ws = (char*)d_ws;
  size_t off = 0;
  auto alloc = [&](size_t bytes) { void* p = ws + off; off += (bytes + 255) & ~(size_t)255; return p; };

  __bf16* xb  = (__bf16*)alloc(NX * 2);     // also reused for attention output H
  __bf16* wqb = (__bf16*)alloc(NW * 2);
  __bf16* wkb = (__bf16*)alloc(NW * 2);
  __bf16* wvb = (__bf16*)alloc(NW * 2);
  __bf16* wob = (__bf16*)alloc(NW * 2);
  __bf16* Qb  = (__bf16*)alloc(NX * 2);
  __bf16* Kb  = (__bf16*)alloc(NX * 2);
  __bf16* Vb  = (__bf16*)alloc(NX * 2);
  __bf16* Vt  = (__bf16*)alloc(NX * 2);
  __bf16* Hb  = xb;                         // alias: x no longer needed after projections

  const int M = B_ * S_, N = D_, Kdim = D_;

  // 1) fp32 -> bf16
  cvt_f32_to_bf16<<<(int)((NX + 255) / 256), 256, 0, stream>>>(x, xb, (int)NX);
  cvt_f32_to_bf16<<<(int)((NW + 255) / 256), 256, 0, stream>>>(wq, wqb, (int)NW);
  cvt_f32_to_bf16<<<(int)((NW + 255) / 256), 256, 0, stream>>>(wk, wkb, (int)NW);
  cvt_f32_to_bf16<<<(int)((NW + 255) / 256), 256, 0, stream>>>(wv, wvb, (int)NW);
  cvt_f32_to_bf16<<<(int)((NW + 255) / 256), 256, 0, stream>>>(wo, wob, (int)NW);

  // 2) Q/K/V projections (WMMA bf16, TDM-staged weight tiles)
  dim3 gg(N / 64, M / 256);
  gemm_xwT<__bf16><<<gg, 256, 0, stream>>>(xb, wqb, Qb, M, N, Kdim);
  gemm_xwT<__bf16><<<gg, 256, 0, stream>>>(xb, wkb, Kb, M, N, Kdim);
  gemm_xwT<__bf16><<<gg, 256, 0, stream>>>(xb, wvb, Vb, M, N, Kdim);

  // 3) RoPE on Q and K
  const int nrope = B_ * S_ * H_ * 32;
  rope_kernel<<<(nrope + 255) / 256, 256, 0, stream>>>(Qb, pos, nrope);
  rope_kernel<<<(nrope + 255) / 256, 256, 0, stream>>>(Kb, pos, nrope);

  // 4) V -> V^T (B,H,dk,S) for contiguous P*V B-matrix loads
  transpose_v<<<(int)((NX + 255) / 256), 256, 0, stream>>>(Vb, Vt, (int)NX);

  // 5) causal flash attention (WMMA bf16)
  dim3 fg(S_ / 128, H_, B_);
  flash_attn<<<fg, 256, 0, stream>>>(Qb, Kb, Vt, Hb);

  // 6) output projection -> fp32 d_out
  gemm_xwT<float><<<gg, 256, 0, stream>>>(Hb, wob, out, M, N, Kdim);
}
